// ExpertMLPsV2_18013092840056
// MI455X (gfx1250) — matrix operand order repaired and checked by reading, not compiled, not run on hardware
//
#include <hip/hip_runtime.h>
#include <hip/hip_bf16.h>
#include <stdint.h>

// Problem constants
constexpr int kT = 4096;
constexpr int kH = 2048;
constexpr int kI = 4096;
constexpr int kE = 8;
constexpr int kTopK = 2;

// GEMM tiling
constexpr int TM = 128;   // rows (gathered tokens) per workgroup
constexpr int TN = 64;    // output columns per workgroup
constexpr int KB = 32;    // K per WMMA step (bf16)

typedef __attribute__((ext_vector_type(16))) __bf16 bf16x16;
typedef __attribute__((ext_vector_type(8)))  float  f32x8;

union ABFrag {
  uint32_t u[8];
  uint4    q[2];
  bf16x16  v;
};

__device__ __forceinline__ uint16_t f2bf(float x) {
  uint32_t u = __float_as_uint(x);
  u += 0x7fffu + ((u >> 16) & 1u);   // round-to-nearest-even
  return (uint16_t)(u >> 16);
}
__device__ __forceinline__ uint32_t f2bf2(float lo, float hi) {
  return (uint32_t)f2bf(lo) | ((uint32_t)f2bf(hi) << 16);
}
__device__ __forceinline__ f32x8 zero8() {
  f32x8 z = {0.f, 0.f, 0.f, 0.f, 0.f, 0.f, 0.f, 0.f};
  return z;
}
// silu via hardware v_rcp_f32 (avoids IEEE div expansion)
__device__ __forceinline__ float silu_fast(float g) {
  return g * __builtin_amdgcn_rcpf(1.f + __expf(-g));
}

// ---------------------------------------------------------------- utilities
__global__ void moe_zero(float* __restrict__ p, int n) {
  for (int i = blockIdx.x * blockDim.x + threadIdx.x; i < n;
       i += gridDim.x * blockDim.x)
    p[i] = 0.f;
}

__global__ void moe_cvt(const float* __restrict__ x, uint32_t* __restrict__ xb,
                        int n2) {
  for (int i = blockIdx.x * blockDim.x + threadIdx.x; i < n2;
       i += gridDim.x * blockDim.x)
    xb[i] = f2bf2(x[2 * i], x[2 * i + 1]);
}

// masked + L1-normalized affinities, (T,E)
__global__ void moe_aff(const float* __restrict__ aff,
                        const int* __restrict__ eidx,
                        float* __restrict__ affn) {
  int t = blockIdx.x * blockDim.x + threadIdx.x;
  if (t >= kT) return;
  int i0 = eidx[2 * t + 0];
  int i1 = eidx[2 * t + 1];
  float v[kE];
  float s = 0.f;
#pragma unroll
  for (int e = 0; e < kE; ++e) {
    bool sel = (i0 == e) || (i1 == e);
    float x = sel ? aff[t * kE + e] : 0.f;
    v[e] = x;
    s += fabsf(x);
  }
  float inv = 1.f / fmaxf(s, 1e-12f);
#pragma unroll
  for (int e = 0; e < kE; ++e) affn[t * kE + e] = v[e] * inv;
}

// deterministic per-expert token-list compaction (single workgroup)
__global__ __launch_bounds__(1024) void moe_lists(const int* __restrict__ eidx,
                                                  int* __restrict__ lists,
                                                  int* __restrict__ counts,
                                                  int* __restrict__ offs) {
  __shared__ int wsum[32];
  __shared__ int woff[32];
  __shared__ int chunkTot;
  __shared__ int carry;
  const int tid = threadIdx.x;
  const int lane = tid & 31;
  const int wid = tid >> 5;

  for (int e = 0; e < kE; ++e) {
    if (tid == 0) carry = 0;
    __syncthreads();
    for (int c = 0; c < kT / 1024; ++c) {
      int t = c * 1024 + tid;
      int i0 = eidx[2 * t + 0];
      int i1 = eidx[2 * t + 1];
      bool f = (i0 == e) || (i1 == e);
      unsigned long long m = __ballot(f);
      int pre = __popcll(m & ((1ull << lane) - 1ull));
      int tot = __popcll(m);
      if (lane == 0) wsum[wid] = tot;
      __syncthreads();
      if (tid == 0) {
        int run = 0;
        for (int i = 0; i < 32; ++i) { woff[i] = run; run += wsum[i]; }
        chunkTot = run;
      }
      __syncthreads();
      if (f) lists[e * kT + carry + woff[wid] + pre] = t;
      __syncthreads();
      if (tid == 0) carry += chunkTot;
      __syncthreads();
    }
    if (tid == 0) counts[e] = carry;
    __syncthreads();
  }
  if (tid == 0) {
    offs[0] = 0;
    for (int i = 0; i < kE; ++i) offs[i + 1] = offs[i] + counts[i];
  }
}

// ------------------------------------------------------- stage 1: gate & up
// C[m, n] over gathered tokens; h = silu(X Wg) * (X Wu) -> bf16 hbuf
__global__ __launch_bounds__(256) void moe_gemm1(
    const uint16_t* __restrict__ Xb,       // T x H bf16
    const float* __restrict__ Wg,          // E x H x I
    const float* __restrict__ Wu,          // E x H x I
    const int* __restrict__ lists, const int* __restrict__ counts,
    const int* __restrict__ offs,
    uint16_t* __restrict__ hbuf) {         // (sum counts) x I bf16
  const int e = blockIdx.z;
  const int cnt = counts[e];
  const int m0 = blockIdx.y * TM;
  if (m0 >= cnt) return;
  const int n0 = blockIdx.x * TN;
  const long wbase = (long)e * kH * kI;
  const int hrow0 = offs[e] + m0;

  __shared__ __align__(16) uint16_t As[TM * KB];    // [row][k]
  __shared__ __align__(16) uint16_t Bgs[TN * KB];   // [n][k]
  __shared__ __align__(16) uint16_t Bus[TN * KB];   // [n][k]
  __shared__ int rowTok[TM];

  const int tid = threadIdx.x;
  if (tid < TM) {
    int r = m0 + tid;
    rowTok[tid] = (r < cnt) ? lists[e * kT + r] : 0;
  }
  __syncthreads();

  const int lane = tid & 31;
  const int wv = tid >> 5;
  const int mi = wv & 3;   // 4 waves along M (32 rows each)
  const int ni = wv >> 2;  // 2 waves along N (32 cols each)

  f32x8 ag[2][2], au[2][2];
#pragma unroll
  for (int a = 0; a < 2; ++a)
#pragma unroll
    for (int b = 0; b < 2; ++b) { ag[a][b] = zero8(); au[a][b] = zero8(); }

  // A staging: 2 threads per row, 32B each
  const int ar = tid >> 1;
  const int ak = (tid & 1) * 16;
  const uint16_t* xrow = Xb + (long)rowTok[ar] * kH + ak;

  // B staging: 64 n-lanes x 4 k-quads, 2 reps cover k 0..31
  const int bn = tid & 63;
  const int bk = (tid >> 6) * 4;

  for (int k0 = 0; k0 < kH; k0 += KB) {
    {
      const uint4* s = (const uint4*)(xrow + k0);
      uint4* d = (uint4*)(&As[ar * KB + ak]);
      d[0] = s[0];
      d[1] = s[1];
    }
#pragma unroll
    for (int rep = 0; rep < 2; ++rep) {
      int kk = bk + rep * 16;
      const float* g = Wg + wbase + (long)(k0 + kk) * kI + (n0 + bn);
      const float* u = Wu + wbase + (long)(k0 + kk) * kI + (n0 + bn);
      uint2 pg, pu;
      pg.x = f2bf2(g[0], g[kI]);
      pg.y = f2bf2(g[2L * kI], g[3L * kI]);
      pu.x = f2bf2(u[0], u[kI]);
      pu.y = f2bf2(u[2L * kI], u[3L * kI]);
      *(uint2*)(&Bgs[bn * KB + kk]) = pg;
      *(uint2*)(&Bus[bn * KB + kk]) = pu;
    }
    __syncthreads();

    ABFrag a[2], bg[2], bu[2];
#pragma unroll
    for (int ms = 0; ms < 2; ++ms) {
      int row = mi * 32 + ms * 16 + (lane & 15);
      int ko = (lane >> 4) * 16;  // lanes 0-15: K0-15; 16-31: K16-31
      const uint4* p = (const uint4*)(&As[row * KB + ko]);
      a[ms].q[0] = p[0];
      a[ms].q[1] = p[1];
    }
#pragma unroll
    for (int ns = 0; ns < 2; ++ns) {
      int col = ni * 32 + ns * 16 + (lane & 15);
      int ko = (lane >> 4) * 16;  // lanes 0-15: K0-15; 16-31: K16-31
      const uint4* pg = (const uint4*)(&Bgs[col * KB + ko]);
      bg[ns].q[0] = pg[0];
      bg[ns].q[1] = pg[1];
      const uint4* pu = (const uint4*)(&Bus[col * KB + ko]);
      bu[ns].q[0] = pu[0];
      bu[ns].q[1] = pu[1];
    }
#pragma unroll
    for (int ms = 0; ms < 2; ++ms)
#pragma unroll
      for (int ns = 0; ns < 2; ++ns) {
        ag[ms][ns] = __builtin_amdgcn_wmma_f32_16x16x32_bf16(
            false, a[ms].v, false, bg[ns].v, (short)0, ag[ms][ns], false, false);
        au[ms][ns] = __builtin_amdgcn_wmma_f32_16x16x32_bf16(
            false, a[ms].v, false, bu[ns].v, (short)0, au[ms][ns], false, false);
      }
    __syncthreads();
  }

  // epilogue: silu(g)*u -> bf16 hbuf
#pragma unroll
  for (int ms = 0; ms < 2; ++ms)
#pragma unroll
    for (int ns = 0; ns < 2; ++ns) {
      int colg = n0 + ni * 32 + ns * 16 + (lane & 15);
      int rbase = mi * 32 + ms * 16 + ((lane >> 4) << 3);
#pragma unroll
      for (int r = 0; r < 8; ++r) {
        int rt = rbase + r;
        if (m0 + rt < cnt) {
          float hv = silu_fast(ag[ms][ns][r]) * au[ms][ns][r];
          hbuf[(long)(hrow0 + rt) * kI + colg] = f2bf(hv);
        }
      }
    }
}

// ------------------------------------------------ stage 2: down projection
__global__ __launch_bounds__(256) void moe_gemm2(
    const uint16_t* __restrict__ hbuf,     // rows x I bf16
    const float* __restrict__ Wd,          // E x I x H
    const int* __restrict__ lists, const int* __restrict__ counts,
    const int* __restrict__ offs, const float* __restrict__ affn,
    float* __restrict__ out) {             // T x H f32
  const int e = blockIdx.z;
  const int cnt = counts[e];
  const int m0 = blockIdx.y * TM;
  if (m0 >= cnt) return;
  const int n0 = blockIdx.x * TN;
  const long wbase = (long)e * kI * kH;
  const int hrow0 = offs[e] + m0;

  __shared__ __align__(16) uint16_t As[TM * KB];
  __shared__ __align__(16) uint16_t Bs[TN * KB];
  __shared__ int rowTok[TM];
  __shared__ float rowScale[TM];

  const int tid = threadIdx.x;
  if (tid < TM) {
    int r = m0 + tid;
    int tok = (r < cnt) ? lists[e * kT + r] : 0;
    rowTok[tid] = tok;
    rowScale[tid] = (r < cnt) ? affn[tok * kE + e] : 0.f;
  }
  __syncthreads();

  const int lane = tid & 31;
  const int wv = tid >> 5;
  const int mi = wv & 3;
  const int ni = wv >> 2;

  f32x8 acc[2][2];
#pragma unroll
  for (int a = 0; a < 2; ++a)
#pragma unroll
    for (int b = 0; b < 2; ++b) acc[a][b] = zero8();

  const int ar = tid >> 1;
  const int ak = (tid & 1) * 16;
  const bool avalid = (m0 + ar) < cnt;
  const uint16_t* hr = hbuf + (long)(hrow0 + ar) * kI + ak;

  const int bn = tid & 63;
  const int bk = (tid >> 6) * 4;

  for (int k0 = 0; k0 < kI; k0 += KB) {
    {
      uint4* d = (uint4*)(&As[ar * KB + ak]);
      if (avalid) {
        const uint4* s = (const uint4*)(hr + k0);
        d[0] = s[0];
        d[1] = s[1];
      } else {
        uint4 z = {0u, 0u, 0u, 0u};
        d[0] = z;
        d[1] = z;
      }
    }
#pragma unroll
    for (int rep = 0; rep < 2; ++rep) {
      int kk = bk + rep * 16;
      const float* w = Wd + wbase + (long)(k0 + kk) * kH + (n0 + bn);
      uint2 p;
      p.x = f2bf2(w[0], w[kH]);
      p.y = f2bf2(w[2L * kH], w[3L * kH]);
      *(uint2*)(&Bs[bn * KB + kk]) = p;
    }
    __syncthreads();

    ABFrag a[2], b[2];
#pragma unroll
    for (int ms = 0; ms < 2; ++ms) {
      int row = mi * 32 + ms * 16 + (lane & 15);
      int ko = (lane >> 4) * 16;
      const uint4* p = (const uint4*)(&As[row * KB + ko]);
      a[ms].q[0] = p[0];
      a[ms].q[1] = p[1];
    }
#pragma unroll
    for (int ns = 0; ns < 2; ++ns) {
      int col = ni * 32 + ns * 16 + (lane & 15);
      int ko = (lane >> 4) * 16;
      const uint4* p = (const uint4*)(&Bs[col * KB + ko]);
      b[ns].q[0] = p[0];
      b[ns].q[1] = p[1];
    }
#pragma unroll
    for (int ms = 0; ms < 2; ++ms)
#pragma unroll
      for (int ns = 0; ns < 2; ++ns)
        acc[ms][ns] = __builtin_amdgcn_wmma_f32_16x16x32_bf16(
            false, a[ms].v, false, b[ns].v, (short)0, acc[ms][ns], false, false);
    __syncthreads();
  }

  // epilogue: scale by affinity, scatter-accumulate (<=2 addends/elem -> deterministic)
#pragma unroll
  for (int ms = 0; ms < 2; ++ms)
#pragma unroll
    for (int ns = 0; ns < 2; ++ns) {
      int colg = n0 + ni * 32 + ns * 16 + (lane & 15);
      int rbase = mi * 32 + ms * 16 + ((lane >> 4) << 3);
#pragma unroll
      for (int r = 0; r < 8; ++r) {
        int rt = rbase + r;
        if (m0 + rt < cnt) {
          float v = acc[ms][ns][r] * rowScale[rt];
          unsafeAtomicAdd(&out[(long)rowTok[rt] * kH + colg], v);
        }
      }
    }
}

// ------------------------------------------------------------------ launch
extern "C" void kernel_launch(void* const* d_in, const int* in_sizes, int n_in,
                              void* d_out, int out_size, void* d_ws,
                              size_t ws_size, hipStream_t stream) {
  const float* x    = (const float*)d_in[0];  // hidden_states (T,H)
  const float* aff  = (const float*)d_in[1];  // expert_affinities (T,E)
  const int*   eidx = (const int*)d_in[2];    // expert_index (T,TOPK)
  const float* wg   = (const float*)d_in[3];  // (E,H,I)
  const float* wu   = (const float*)d_in[4];  // (E,H,I)
  const float* wd   = (const float*)d_in[5];  // (E,I,H)
  float* out = (float*)d_out;

  char* ws = (char*)d_ws;
  size_t off = 0;
  float* affn = (float*)(ws + off);     off += (size_t)kT * kE * 4;       // 128 KB
  uint16_t* Xb = (uint16_t*)(ws + off); off += (size_t)kT * kH * 2;       // 16 MB
  int* lists = (int*)(ws + off);        off += (size_t)kE * kT * 4;       // 128 KB
  int* counts = (int*)(ws + off);       off += 64;
  int* offs = (int*)(ws + off);         off += 64;
  off = (off + 255) & ~(size_t)255;
  uint16_t* hbuf = (uint16_t*)(ws + off);  // (T*TOPK) x I bf16 = 64 MB

  moe_zero<<<1024, 256, 0, stream>>>(out, kT * kH);
  moe_cvt<<<1024, 256, 0, stream>>>(x, (uint32_t*)Xb, kT * kH / 2);
  moe_aff<<<kT / 256, 256, 0, stream>>>(aff, eidx, affn);
  moe_lists<<<1, 1024, 0, stream>>>(eidx, lists, counts, offs);
  moe_gemm1<<<dim3(kI / TN, kT / TM, kE), 256, 0, stream>>>(
      Xb, wg, wu, lists, counts, offs, hbuf);
  moe_gemm2<<<dim3(kH / TN, kT / TM, kE), 256, 0, stream>>>(
      hbuf, wd, lists, counts, offs, affn, out);
}
